// KeyPointNet_29935922053518
// MI455X (gfx1250) — matrix operand order repaired and checked by reading, not compiled, not run on hardware
//
#include <hip/hip_runtime.h>
#include <hip/hip_bf16.h>

// ---------------- problem constants ----------------
static constexpr int B   = 16;
static constexpr int N   = 2048;
static constexpr int D   = 512;
static constexpr int KP  = 768;           // NUM_KEYPOINTS
static constexpr int BSZ = 128;           // tile edge (both s and t)
static constexpr int KC  = 32;            // bf16 wmma K per step
static constexpr int PITCH = 40;          // LDS tile pitch in halfwords (80B rows: 16B-aligned, bank-friendly)

typedef unsigned short u16;
typedef __attribute__((ext_vector_type(16))) __bf16 v16bf;
typedef __attribute__((ext_vector_type(8)))  float  v8f;

union FragB16 { uint4 q[2]; v16bf v; };

__device__ __forceinline__ u16 f2bf(float f) {
  unsigned u = __float_as_uint(f);
  u += 0x7fffu + ((u >> 16) & 1u);       // round-to-nearest-even
  return (u16)(u >> 16);
}

// order-preserving float -> uint map (monotone): enables uint atomicMax on scores
__device__ __forceinline__ unsigned fkey(float f) {
  unsigned u = __float_as_uint(f);
  return (u & 0x80000000u) ? ~u : (u | 0x80000000u);
}

// ---------------- 0) zero the score-key arrays -------------------------------------
__global__ void zero_keys(unsigned* __restrict__ k) {
  k[blockIdx.x * 256 + threadIdx.x] = 0u;
}

// ---------------- 1) inverse column norms over D -----------------------------------
__global__ void col_norm(const float* __restrict__ eb, float* __restrict__ inv) {
  int g = blockIdx.x * 256 + threadIdx.x;      // [0, B*N)
  int b = g >> 11, n = g & (N - 1);
  const float* p = eb + (size_t)b * D * N + n;
  float s = 0.f;
  #pragma unroll 8
  for (int d = 0; d < D; ++d) { float v = p[(size_t)d * N]; s += v * v; }
  inv[g] = 1.0f / fmaxf(sqrtf(s), 1e-12f);
}

// ---------------- 1b) normalize + convert + transpose to bf16 [b][n][d] ------------
// 64x64 tiles via LDS; coalesced f32 reads (n-contig) and uint writes (d-contig).
__global__ __launch_bounds__(256) void tconv(const float* __restrict__ eb,
                                             const float* __restrict__ inv,
                                             u16* __restrict__ outT) {
  __shared__ u16 T[64 * 66];                    // pitch 66 halfwords = 132B rows (4B aligned)
  const int n0 = blockIdx.x * 64;
  const int d0 = blockIdx.y * 64;
  const int b  = blockIdx.z;
  const int tid = threadIdx.x;
  #pragma unroll
  for (int i = 0; i < 16; ++i) {
    int li = i * 256 + tid;
    int dl = li >> 6, nl = li & 63;
    float v = eb[((size_t)b * D + d0 + dl) * N + n0 + nl] * inv[b * N + n0 + nl];
    T[nl * 66 + dl] = f2bf(v);
  }
  __syncthreads();
  const unsigned* Tu = (const unsigned*)T;      // 33 uints per row
  #pragma unroll
  for (int i = 0; i < 8; ++i) {
    int li = i * 256 + tid;
    int nl = li >> 5, du = li & 31;             // du indexes pairs of d
    unsigned* orow = (unsigned*)(outT + ((size_t)b * N + n0 + nl) * D + d0);
    orow[du] = Tu[nl * 33 + du];
  }
}

// ---------------- shared reduction helper (fused row/col max -> atomicMax) ----------
__device__ __forceinline__ void reduce_max_atomic(
    v8f acc[2][4], int b, int s0, int t0, int wS, int wT, int lane,
    unsigned* __restrict__ keyS, unsigned* __restrict__ keyT) {
  const int lr = lane & 15;
  const int h  = lane >> 4;
  // row max (over t): reduce across 16-lane column groups, atomicMax per row
  #pragma unroll
  for (int a = 0; a < 2; ++a) {
    #pragma unroll
    for (int j = 0; j < 8; ++j) {
      float m = fmaxf(fmaxf(acc[a][0][j], acc[a][1][j]),
                      fmaxf(acc[a][2][j], acc[a][3][j]));
      #pragma unroll
      for (int off = 1; off < 16; off <<= 1) m = fmaxf(m, __shfl_xor(m, off, 32));
      if (lr == 0) {
        int row = s0 + wS + a * 16 + j + (h << 3);
        atomicMax(&keyS[b * N + row], fkey(m));
      }
    }
  }
  // col max (over s): per-lane max over 16 rows, merge halves, atomicMax per col
  #pragma unroll
  for (int c = 0; c < 4; ++c) {
    float m = -3.4e38f;
    #pragma unroll
    for (int a = 0; a < 2; ++a)
      #pragma unroll
      for (int j = 0; j < 8; ++j) m = fmaxf(m, acc[a][c][j]);
    m = fmaxf(m, __shfl_xor(m, 16, 32));
    if (lane < 16) {
      int col = t0 + wT + c * 16 + lane;
      atomicMax(&keyT[b * N + col], fkey(m));
    }
  }
}

// ---------------- 2a) bf16 WMMA GEMM, async global->LDS staging, double-buffered ----
// Stage one 128x32 bf16 tile (row n: 64B contiguous) as 512 x 16B async packets.
__device__ __forceinline__ void stage_tile(const char* sbase, unsigned ldsbase,
                                           int k0, int tid) {
  #pragma unroll
  for (int p = tid; p < 512; p += 256) {
    int row = p >> 2;
    int seg = p & 3;
    unsigned lds  = ldsbase + (unsigned)(row * (PITCH * 2) + seg * 16);
    unsigned voff = (unsigned)(row * (D * 2) + k0 * 2 + seg * 16);
    asm volatile("global_load_async_to_lds_b128 %0, %1, %2"
                 :: "v"(lds), "v"(voff), "s"(sbase)
                 : "memory");
  }
}

__global__ __launch_bounds__(256) void gemm_max_bf16(
    const u16* __restrict__ fS, const u16* __restrict__ fT,
    unsigned* __restrict__ keyS, unsigned* __restrict__ keyT) {
  __shared__ __align__(16) u16 As[2][BSZ * PITCH];
  __shared__ __align__(16) u16 Bs[2][BSZ * PITCH];

  const int b  = blockIdx.z;
  const int s0 = blockIdx.x * BSZ;
  const int t0 = blockIdx.y * BSZ;
  const int tid  = threadIdx.x;
  const int wave = tid >> 5;
  const int lane = tid & 31;
  const int wS = (wave & 3) * 32;
  const int wT = (wave >> 2) * 64;
  const int lr = lane & 15;
  const int h  = lane >> 4;

  const char* sbA = (const char*)(fS + ((size_t)b * N + s0) * D);
  const char* sbB = (const char*)(fT + ((size_t)b * N + t0) * D);
  unsigned ldsA[2] = { (unsigned)(uintptr_t)&As[0][0], (unsigned)(uintptr_t)&As[1][0] };
  unsigned ldsB[2] = { (unsigned)(uintptr_t)&Bs[0][0], (unsigned)(uintptr_t)&Bs[1][0] };

  v8f acc[2][4];
  #pragma unroll
  for (int a = 0; a < 2; ++a)
    #pragma unroll
    for (int c = 0; c < 4; ++c) acc[a][c] = (v8f)0.f;

  stage_tile(sbA, ldsA[0], 0, tid);
  stage_tile(sbB, ldsB[0], 0, tid);
  asm volatile("s_wait_asynccnt 0x0" ::: "memory");
  __syncthreads();

  for (int kc = 0; kc < D / KC; ++kc) {
    const int cur = kc & 1;
    if (kc + 1 < D / KC) {                  // prefetch next chunk into other buffer
      stage_tile(sbA, ldsA[cur ^ 1], (kc + 1) * KC, tid);
      stage_tile(sbB, ldsB[cur ^ 1], (kc + 1) * KC, tid);
    }
    const uint4* A4 = (const uint4*)As[cur];   // 5 uint4 per LDS row
    const uint4* B4 = (const uint4*)Bs[cur];

    FragB16 bf[4];
    #pragma unroll
    for (int c = 0; c < 4; ++c) {
      int r = wT + c * 16 + lr;
      bf[c].q[0] = B4[r * 5 + 2 * h];
      bf[c].q[1] = B4[r * 5 + 2 * h + 1];
    }
    #pragma unroll
    for (int a = 0; a < 2; ++a) {
      FragB16 af;
      int r = wS + a * 16 + lr;
      af.q[0] = A4[r * 5 + 2 * h];
      af.q[1] = A4[r * 5 + 2 * h + 1];
      #pragma unroll
      for (int c = 0; c < 4; ++c) {
        acc[a][c] = __builtin_amdgcn_wmma_f32_16x16x32_bf16(
            false, af.v, false, bf[c].v, (short)0, acc[a][c], false, false);
      }
    }
    asm volatile("s_wait_asynccnt 0x0" ::: "memory");  // own async writes landed
    __syncthreads();                                    // everyone's landed
  }

  reduce_max_atomic(acc, b, s0, t0, wS, wT, lane, keyS, keyT);
}

// ---------------- 2b) fallback: f32-staged WMMA GEMM (small workspace) --------------
__global__ __launch_bounds__(256) void gemm_max(
    const float* __restrict__ seb, const float* __restrict__ teb,
    const float* __restrict__ invs, const float* __restrict__ invt,
    unsigned* __restrict__ keyS, unsigned* __restrict__ keyT) {
  __shared__ __align__(16) u16 As[BSZ * PITCH];
  __shared__ __align__(16) u16 Bs[BSZ * PITCH];
  __shared__ float invA[BSZ], invB[BSZ];

  const int b  = blockIdx.z;
  const int s0 = blockIdx.x * BSZ;
  const int t0 = blockIdx.y * BSZ;
  const int tid = threadIdx.x;

  if (tid < BSZ)            invA[tid]       = invs[b * N + s0 + tid];
  else if (tid < 2 * BSZ)   invB[tid - BSZ] = invt[b * N + t0 + (tid - BSZ)];
  __syncthreads();

  const int wave = tid >> 5;
  const int lane = tid & 31;
  const int wS = (wave & 3) * 32;
  const int wT = (wave >> 2) * 64;
  const int lr = lane & 15;
  const int h  = lane >> 4;

  v8f acc[2][4];
  #pragma unroll
  for (int a = 0; a < 2; ++a)
    #pragma unroll
    for (int c = 0; c < 4; ++c) acc[a][c] = (v8f)0.f;

  const float* Ag = seb + (size_t)b * D * N + s0;
  const float* Bg = teb + (size_t)b * D * N + t0;
  const uint4* A4 = (const uint4*)As;
  const uint4* B4 = (const uint4*)Bs;

  for (int k0 = 0; k0 < D; k0 += KC) {
    #pragma unroll
    for (int i = 0; i < 16; ++i) {
      int li = i * 256 + tid;
      int dl = li >> 7;
      int nl = li & 127;
      As[nl * PITCH + dl] = f2bf(Ag[(size_t)(k0 + dl) * N + nl] * invA[nl]);
      Bs[nl * PITCH + dl] = f2bf(Bg[(size_t)(k0 + dl) * N + nl] * invB[nl]);
    }
    __syncthreads();

    FragB16 bf[4];
    #pragma unroll
    for (int c = 0; c < 4; ++c) {
      int r = wT + c * 16 + lr;
      bf[c].q[0] = B4[r * 5 + 2 * h];
      bf[c].q[1] = B4[r * 5 + 2 * h + 1];
    }
    #pragma unroll
    for (int a = 0; a < 2; ++a) {
      FragB16 af;
      int r = wS + a * 16 + lr;
      af.q[0] = A4[r * 5 + 2 * h];
      af.q[1] = A4[r * 5 + 2 * h + 1];
      #pragma unroll
      for (int c = 0; c < 4; ++c) {
        acc[a][c] = __builtin_amdgcn_wmma_f32_16x16x32_bf16(
            false, af.v, false, bf[c].v, (short)0, acc[a][c], false, false);
      }
    }
    __syncthreads();
  }

  reduce_max_atomic(acc, b, s0, t0, wS, wT, lane, keyS, keyT);
}

// ---------------- 3) per-(batch,side) bitonic top-k ----------------------------------
__global__ __launch_bounds__(1024) void topk_sort(
    const unsigned* __restrict__ keyS, const unsigned* __restrict__ keyT,
    int* __restrict__ idxS, int* __restrict__ idxT) {
  __shared__ unsigned long long sk[N];
  const int b    = blockIdx.x & 15;
  const int side = blockIdx.x >> 4;
  const unsigned* key = side ? keyT : keyS;
  int* idx = side ? idxT : idxS;
  const int tid = threadIdx.x;

  for (int i = tid; i < N; i += 1024)
    sk[i] = ((unsigned long long)key[b * N + i] << 32) | (unsigned)(N - 1 - i);
  __syncthreads();

  for (int k = 2; k <= N; k <<= 1) {
    for (int j = k >> 1; j > 0; j >>= 1) {
      for (int t = tid; t < N; t += 1024) {
        int ixj = t ^ j;
        if (ixj > t) {
          bool desc = ((t & k) == 0);
          unsigned long long x = sk[t], y = sk[ixj];
          if (desc ? (x < y) : (x > y)) { sk[t] = y; sk[ixj] = x; }
        }
      }
      __syncthreads();
    }
  }
  if (tid < KP) idx[b * KP + tid] = (N - 1) - (int)(sk[tid] & 0xFFFFFFFFull);
}

// ---------------- 4) gathers ----------------------------------------------------------
__global__ void gather_pts(const float* __restrict__ src, const float* __restrict__ tgt,
                           const int* __restrict__ idxS, const int* __restrict__ idxT,
                           float* __restrict__ out) {
  int g = blockIdx.x * 256 + threadIdx.x;            // [0, 2*B*KP)
  int side = g / (B * KP);
  int r = g - side * (B * KP);
  int b = r / KP, k = r - b * KP;
  const int*   idx = side ? idxT : idxS;
  const float* pts = side ? tgt  : src;
  int n = idx[b * KP + k];
  float*       o = out + (size_t)side * (B * KP * 3) + (size_t)r * 3;
  const float* p = pts + ((size_t)b * N + n) * 3;
  o[0] = p[0]; o[1] = p[1]; o[2] = p[2];
}

__global__ void gather_eb(const float* __restrict__ seb, const float* __restrict__ teb,
                          const int* __restrict__ idxS, const int* __restrict__ idxT,
                          float* __restrict__ out) {
  long long g = (long long)blockIdx.x * 256 + threadIdx.x;   // [0, 2*B*D*KP)
  const long long per = (long long)B * D * KP;
  int side = (int)(g / per);
  long long r = g - (long long)side * per;
  int b   = (int)(r / ((long long)D * KP));
  int rem = (int)(r - (long long)b * D * KP);
  int d = rem / KP, k = rem - d * KP;
  const int*   idx = side ? idxT : idxS;
  const float* eb  = side ? teb  : seb;
  int n = idx[b * KP + k];
  out[(size_t)side * per + (size_t)r] = eb[((size_t)b * D + d) * N + n];
}

// ---------------- launcher ------------------------------------------------------------
extern "C" void kernel_launch(void* const* d_in, const int* in_sizes, int n_in,
                              void* d_out, int out_size, void* d_ws, size_t ws_size,
                              hipStream_t stream) {
  (void)in_sizes; (void)n_in; (void)out_size;
  const float* src    = (const float*)d_in[0];
  const float* tgt    = (const float*)d_in[1];
  const float* src_eb = (const float*)d_in[4];
  const float* tgt_eb = (const float*)d_in[5];
  float* out = (float*)d_out;

  // workspace layout
  char* ws = (char*)d_ws;
  float*    inv_src = (float*)ws;                       // B*N
  float*    inv_tgt = inv_src + B * N;                  // B*N
  unsigned* key_src = (unsigned*)(inv_tgt + B * N);     // B*N (contiguous with key_tgt)
  unsigned* key_tgt = key_src + B * N;                  // B*N
  int*      idx_src = (int*)(key_tgt + B * N);          // B*KP
  int*      idx_tgt = idx_src + B * KP;                 // B*KP
  u16*      featS   = (u16*)(idx_tgt + B * KP);         // B*N*D bf16 (optional)
  u16*      featT   = featS + (size_t)B * N * D;

  const size_t base_need = (size_t)(4 * B * N) * 4 + (size_t)(2 * B * KP) * 4;
  const size_t feat_need = 2 * (size_t)B * N * D * sizeof(u16);
  const bool use_bf16_path = (ws_size >= base_need + feat_need);

  // 0) init score keys
  zero_keys<<<(2 * B * N) / 256, 256, 0, stream>>>(key_src);

  // 1) inverse L2 norms
  col_norm<<<(B * N) / 256, 256, 0, stream>>>(src_eb, inv_src);
  col_norm<<<(B * N) / 256, 256, 0, stream>>>(tgt_eb, inv_tgt);

  // 2) WMMA score GEMM with fused row/col max
  if (use_bf16_path) {
    dim3 tg(N / 64, D / 64, B);
    tconv<<<tg, 256, 0, stream>>>(src_eb, inv_src, featS);
    tconv<<<tg, 256, 0, stream>>>(tgt_eb, inv_tgt, featT);
    gemm_max_bf16<<<dim3(N / BSZ, N / BSZ, B), 256, 0, stream>>>(
        featS, featT, key_src, key_tgt);
  } else {
    gemm_max<<<dim3(N / BSZ, N / BSZ, B), 256, 0, stream>>>(
        src_eb, tgt_eb, inv_src, inv_tgt, key_src, key_tgt);
  }

  // 3) top-k indices
  topk_sort<<<32, 1024, 0, stream>>>(key_src, key_tgt, idx_src, idx_tgt);

  // 4) gathers into concatenated output
  gather_pts<<<(2 * B * KP) / 256, 256, 0, stream>>>(src, tgt, idx_src, idx_tgt, out);
  gather_eb<<<(int)((2LL * B * D * KP) / 256), 256, 0, stream>>>(
      src_eb, tgt_eb, idx_src, idx_tgt, out + 2 * B * KP * 3);
}
